// LRCP_split_24077586661555
// MI455X (gfx1250) — matrix operand order repaired and checked by reading, not compiled, not run on hardware
//
#include <hip/hip_runtime.h>
#include <math.h>

#ifndef __has_builtin
#define __has_builtin(x) 0
#endif

typedef __attribute__((ext_vector_type(16))) __bf16 v16bf;
typedef __attribute__((ext_vector_type(8)))  float  v8f;

#if __has_builtin(__builtin_amdgcn_wmma_f32_16x16x4_f32)
#define HAVE_WF32 1
typedef __attribute__((ext_vector_type(2))) float v2f;
#else
#define HAVE_WF32 0
#endif

union FragBF { v16bf v; unsigned u[8]; };

__device__ __forceinline__ unsigned short f2bf(float x) {
  union { float f; unsigned u; } c; c.f = x;
  unsigned r = c.u + 0x7FFFu + ((c.u >> 16) & 1u);
  return (unsigned short)(r >> 16);
}

// Async global->LDS copy (CDNA5, ASYNCcnt-tracked). 16B per lane.
__device__ __forceinline__ void async_g2l_b128(unsigned lds_off, const void* gaddr) {
  asm volatile("global_load_async_to_lds_b128 %0, %1, off"
               :: "v"(lds_off), "v"((unsigned long long)(uintptr_t)gaddr)
               : "memory");
}
#define WAIT_ASYNC(n) asm volatile("s_wait_asynccnt " #n ::: "memory")

// Exact int div/mod via float reciprocal (valid for x < 2^23, one +/-1 fixup).
__device__ __forceinline__ int fdivmod(int x, int d, float rcp, int& rem) {
  int q = (int)((float)x * rcp);
  int r = x - q * d;
  if (r < 0) { --q; r += d; }
  else if (r >= d) { ++q; r -= d; }
  rem = r;
  return q;
}

// fp32 -> bf16 tensor conversion
__global__ void k_f2bf(const float* __restrict__ X, unsigned short* __restrict__ O,
                       long long total) {
  long long idx = (long long)blockIdx.x * 256 + threadIdx.x;
  if (idx < total) O[idx] = f2bf(X[idx]);
}

// ---------------------------------------------------------------------------
// Implicit-GEMM conv (NCHW), bf16 WMMA, 64x64 tile / 128 threads (4 waves).
// Weights async-copied to LDS (overlaps the im2col VALU gather).
// M = OC, N = B*OH*OW, K = IC*KH*KW
// ---------------------------------------------------------------------------
__global__ __launch_bounds__(128)
void k_conv_wmma(const unsigned short* __restrict__ Xbf,
                 const unsigned short* __restrict__ Wbf,
                 float* __restrict__ O,
                 int Bn, int IC, int IH, int IW,
                 int OC, int OH, int OW,
                 int KH, int KW, int stride, int pad,
                 float rcp_khw, float rcp_kw, float rcp_ohw, float rcp_ow) {
  const int Ktot = IC * KH * KW;
  const int Ntot = Bn * OH * OW;
  const int khw = KH * KW, ohw = OH * OW;
  const int Mbase = blockIdx.y * 64;
  const int Nbase = blockIdx.x * 64;
  __shared__ __align__(16) unsigned short As[64 * 40];  // [m][k], 80B rows (16B aligned)
  __shared__ __align__(16) unsigned short Bs[64 * 40];  // [n][k]
  const int tid  = threadIdx.x;
  const int lane = tid & 31, wid = tid >> 5;
  const int wm = wid >> 1, wn = wid & 1;
  const int lm = lane & 15, kg = lane >> 4;
  const bool kasync = ((Ktot & 31) == 0);  // all convs except the 7x7 stem

  v8f acc[2][2];
#pragma unroll
  for (int i = 0; i < 2; ++i)
#pragma unroll
    for (int j = 0; j < 2; ++j)
      acc[i][j] = (v8f){0.f,0.f,0.f,0.f,0.f,0.f,0.f,0.f};

  for (int k0 = 0; k0 < Ktot; k0 += 32) {
    if (kasync) {
      // A tile 64x32 bf16 = 256 x 16B chunks, async DMA into LDS
#pragma unroll
      for (int c = 0; c < 2; ++c) {
        int chunk = c * 128 + tid;
        int m = chunk >> 2, k8 = (chunk & 3) * 8;
        async_g2l_b128((unsigned)(uintptr_t)&As[m * 40 + k8],
                       &Wbf[(long long)(Mbase + m) * Ktot + k0 + k8]);
      }
    } else {
#pragma unroll
      for (int it = 0; it < 16; ++it) {
        int idx = it * 128 + tid;
        int m = idx >> 5, k = idx & 31;
        int gm = Mbase + m, gk = k0 + k;
        As[m * 40 + k] = (gm < OC && gk < Ktot)
                           ? Wbf[(long long)gm * Ktot + gk] : (unsigned short)0;
      }
    }
    // B tile: im2col gather [64 n][32 k] (bf16), overlaps the async weight DMA
#pragma unroll
    for (int it = 0; it < 16; ++it) {
      int idx = it * 128 + tid;
      int n = idx >> 5, k = idx & 31;
      int gn = Nbase + n, gk = k0 + k;
      unsigned short v = 0;
      if (gn < Ntot && gk < Ktot) {
        int rem, kx, sr, ox;
        int ic = fdivmod(gk, khw, rcp_khw, rem);
        int ky = fdivmod(rem, KW, rcp_kw, kx);
        int bb = fdivmod(gn, ohw, rcp_ohw, sr);
        int oy = fdivmod(sr, OW, rcp_ow, ox);
        int iy = oy * stride - pad + ky;
        int ix = ox * stride - pad + kx;
        if (iy >= 0 && iy < IH && ix >= 0 && ix < IW)
          v = Xbf[(((long long)bb * IC + ic) * IH + iy) * IW + ix];
      }
      Bs[n * 40 + k] = v;
    }
    if (kasync) { WAIT_ASYNC(0x0); }
    __syncthreads();
#pragma unroll
    for (int i = 0; i < 2; ++i) {
      FragBF a;
      int ar = 32 * wm + 16 * i + lm;
#pragma unroll
      for (int v = 0; v < 8; ++v) {
        int K0 = ((v < 4) ? 2 * v : 16 + 2 * (v - 4)) + 8 * kg;  // ISA 16-bit A layout
        a.u[v] = *(const unsigned*)&As[ar * 40 + K0];
      }
#pragma unroll
      for (int j = 0; j < 2; ++j) {
        FragBF bf;
        int br = 32 * wn + 16 * j + lm;
#pragma unroll
        for (int v = 0; v < 8; ++v)
          bf.u[v] = *(const unsigned*)&Bs[br * 40 + ((v < 4) ? 2 * v : 16 + 2 * (v - 4)) + 8 * kg];  // ISA 16-bit B layout
        acc[i][j] = __builtin_amdgcn_wmma_f32_16x16x32_bf16(
            false, a.v, false, bf.v, (short)0, acc[i][j], false, false);
      }
    }
    __syncthreads();
  }
  // epilogue: D VGPR r holds row (r + 8*kg), col = lm
#pragma unroll
  for (int i = 0; i < 2; ++i)
#pragma unroll
    for (int j = 0; j < 2; ++j)
#pragma unroll
      for (int r = 0; r < 8; ++r) {
        int gm = Mbase + 32 * wm + 16 * i + r + 8 * kg;
        int gn = Nbase + 32 * wn + 16 * j + lm;
        if (gm < OC && gn < Ntot) {
          int sr, ox;
          int bb = fdivmod(gn, ohw, rcp_ohw, sr);
          int oy = fdivmod(sr, OW, rcp_ow, ox);
          O[(((long long)bb * OC + gm) * OH + oy) * OW + ox] = acc[i][j][r];
        }
      }
}

// ---------------------------------------------------------------------------
// Batched f32 GEMM: C[b] = alpha*A[b]@op(B[b]) + betaDiag*I.
// Requires M,N multiples of 64, K multiple of 16 (true for all call sites).
// Double-buffered async global->LDS pipeline; V_WMMA_F32_16X16X4_F32 math.
// ---------------------------------------------------------------------------
__global__ __launch_bounds__(128)
void k_mm(const float* __restrict__ Ain, const float* __restrict__ Bin,
          float* __restrict__ Cout,
          int M, int N, int K, long long sA, long long sB, long long sC,
          float alpha, float betaDiag, int transB) {
  const float* A = Ain + (long long)blockIdx.z * sA;
  const float* B = Bin + (long long)blockIdx.z * sB;
  float* C = Cout + (long long)blockIdx.z * sC;
  const int Mbase = blockIdx.y * 64, Nbase = blockIdx.x * 64;
  __shared__ __align__(16) float As[2][64 * 20];  // [m][k], 80B rows
  __shared__ __align__(16) float Bs[2][1360];     // transB: [n][k] s20; else [k][n] s68
  const int tid  = threadIdx.x;
  const int lane = tid & 31, wid = tid >> 5;
  const int wm = wid >> 1, wn = wid & 1;
  const int lm = lane & 15, kg = lane >> 4;
  const int nT = K >> 4;

  auto issue = [&](int t, int buf) {
    int k0 = t << 4;
#pragma unroll
    for (int c = 0; c < 2; ++c) {   // A: 64x16 f32 = 256 x 16B chunks
      int chunk = c * 128 + tid;
      int m = chunk >> 2, k4 = (chunk & 3) * 4;
      async_g2l_b128((unsigned)(uintptr_t)&As[buf][m * 20 + k4],
                     &A[(long long)(Mbase + m) * K + k0 + k4]);
    }
    if (transB) {
#pragma unroll
      for (int c = 0; c < 2; ++c) {
        int chunk = c * 128 + tid;
        int n = chunk >> 2, k4 = (chunk & 3) * 4;
        async_g2l_b128((unsigned)(uintptr_t)&Bs[buf][n * 20 + k4],
                       &B[(long long)(Nbase + n) * K + k0 + k4]);
      }
    } else {
#pragma unroll
      for (int c = 0; c < 2; ++c) {  // B tile 16x64 from [K][N]
        int chunk = c * 128 + tid;
        int kk = chunk >> 4, n4 = (chunk & 15) * 4;
        async_g2l_b128((unsigned)(uintptr_t)&Bs[buf][kk * 68 + n4],
                       &B[(long long)(k0 + kk) * N + Nbase + n4]);
      }
    }
  };

#if HAVE_WF32
  v8f acc[2][2];
#pragma unroll
  for (int i = 0; i < 2; ++i)
#pragma unroll
    for (int j = 0; j < 2; ++j)
      acc[i][j] = (v8f){0.f,0.f,0.f,0.f,0.f,0.f,0.f,0.f};
#else
  float accs[32];
#pragma unroll
  for (int r = 0; r < 32; ++r) accs[r] = 0.f;
  const int fn = tid & 63;
  const int frbase = (tid >> 6) * 32;
#endif

  issue(0, 0);
  for (int t = 0; t < nT; ++t) {
    int buf = t & 1;
    if (t + 1 < nT) {
      issue(t + 1, buf ^ 1);   // prefetch next tile into other buffer
      WAIT_ASYNC(0x4);         // wait for tile t (leave t+1's 4 ops in flight)
    } else {
      WAIT_ASYNC(0x0);
    }
    __syncthreads();
#if HAVE_WF32
#pragma unroll
    for (int s = 0; s < 4; ++s) {
#pragma unroll
      for (int i = 0; i < 2; ++i) {
        v2f af;
        int ar = 32 * wm + 16 * i + lm;
        af[0] = As[buf][ar * 20 + 4 * s + 2 * kg + 0];  // f32 A layout: VGPR e -> K=2*kg+e
        af[1] = As[buf][ar * 20 + 4 * s + 2 * kg + 1];
#pragma unroll
        for (int j = 0; j < 2; ++j) {
          v2f bfv;
          int br = 32 * wn + 16 * j + lm;
          if (transB) {
            bfv[0] = Bs[buf][br * 20 + 4 * s + 2 * kg + 0];
            bfv[1] = Bs[buf][br * 20 + 4 * s + 2 * kg + 1];
          } else {
            bfv[0] = Bs[buf][(4 * s + 2 * kg + 0) * 68 + br];
            bfv[1] = Bs[buf][(4 * s + 2 * kg + 1) * 68 + br];
          }
          acc[i][j] = __builtin_amdgcn_wmma_f32_16x16x4_f32(
              false, af, false, bfv, (short)0, acc[i][j], false, false);
        }
      }
    }
#else
    for (int k = 0; k < 16; ++k) {
      float bv = transB ? Bs[buf][fn * 20 + k] : Bs[buf][k * 68 + fn];
#pragma unroll
      for (int r = 0; r < 32; ++r) accs[r] += As[buf][(frbase + r) * 20 + k] * bv;
    }
#endif
    __syncthreads();
  }
#if HAVE_WF32
#pragma unroll
  for (int i = 0; i < 2; ++i)
#pragma unroll
    for (int j = 0; j < 2; ++j)
#pragma unroll
      for (int r = 0; r < 8; ++r) {
        int gm = Mbase + 32 * wm + 16 * i + r + 8 * kg;
        int gn = Nbase + 32 * wn + 16 * j + lm;
        C[(long long)gm * N + gn] = alpha * acc[i][j][r] + (gm == gn ? betaDiag : 0.f);
      }
#else
  for (int r = 0; r < 32; ++r) {
    int gm = Mbase + frbase + r, gn = Nbase + fn;
    C[(long long)gm * N + gn] = alpha * accs[r] + (gm == gn ? betaDiag : 0.f);
  }
#endif
}

// ---------------------------------------------------------------------------
// BatchNorm (batch stats) + fused apply (+residual, +relu)
// ---------------------------------------------------------------------------
__global__ void k_bn_stats(const float* __restrict__ X, float* mean, float* rstd,
                           int Bn, int C, int P) {
  int c = blockIdx.x;
  float s = 0.f, ss = 0.f;
  for (int b = 0; b < Bn; ++b) {
    const float* Xp = X + ((long long)b * C + c) * P;
    for (int i = threadIdx.x; i < P; i += 256) {
      float v = Xp[i];
      s += v; ss += v * v;
    }
  }
  __shared__ float rs[256], rss[256];
  rs[threadIdx.x] = s; rss[threadIdx.x] = ss;
  __syncthreads();
  for (int st = 128; st > 0; st >>= 1) {
    if (threadIdx.x < st) { rs[threadIdx.x] += rs[threadIdx.x + st]; rss[threadIdx.x] += rss[threadIdx.x + st]; }
    __syncthreads();
  }
  if (threadIdx.x == 0) {
    float tot = (float)(Bn * P);
    float m = rs[0] / tot;
    float var = rss[0] / tot - m * m;
    mean[c] = m;
    rstd[c] = rsqrtf(var + 1e-5f);
  }
}

// grid: (ceil(P/256), Bn*C)
__global__ void k_bn_apply(const float* __restrict__ X, float* __restrict__ O,
                           const float* __restrict__ mean, const float* __restrict__ rstd,
                           const float* __restrict__ g, const float* __restrict__ bb,
                           const float* __restrict__ resid, int C, int P, int relu) {
  int i = blockIdx.x * 256 + threadIdx.x;
  if (i >= P) return;
  int bc = blockIdx.y;
  int c = bc % C;  // scalar (uniform per block)
  long long idx = (long long)bc * P + i;
  float v = g[c] * (X[idx] - mean[c]) * rstd[c] + bb[c];
  if (resid) v += resid[idx];
  if (relu) v = fmaxf(v, 0.f);
  O[idx] = v;
}

// grid: (ceil(OH*OW/256), Bn*C)
__global__ void k_maxpool(const float* __restrict__ X, float* __restrict__ O,
                          int IH, int IW, int OH, int OW, float rcp_ow) {
  int i = blockIdx.x * 256 + threadIdx.x;
  if (i >= OH * OW) return;
  int bc = blockIdx.y;
  int ox;
  int oy = fdivmod(i, OW, rcp_ow, ox);
  const float* Xp = X + (long long)bc * IH * IW;
  float m = -3.402823466e38f;
  for (int dy = 0; dy < 3; ++dy) {
    int iy = oy * 2 - 1 + dy;
    if (iy < 0 || iy >= IH) continue;
    for (int dx = 0; dx < 3; ++dx) {
      int ix = ox * 2 - 1 + dx;
      if (ix < 0 || ix >= IW) continue;
      m = fmaxf(m, Xp[iy * IW + ix]);
    }
  }
  O[(long long)bc * OH * OW + i] = m;
}

// mean over spatial; one block per (b,c)
__global__ void k_gap(const float* __restrict__ X, float* __restrict__ O, int P) {
  int bc = blockIdx.x;
  float s = 0.f;
  for (int i = threadIdx.x; i < P; i += blockDim.x) s += X[(long long)bc * P + i];
  __shared__ float red[128];
  red[threadIdx.x] = s; __syncthreads();
  for (int st = 64; st > 0; st >>= 1) {
    if (threadIdx.x < st) red[threadIdx.x] += red[threadIdx.x + st];
    __syncthreads();
  }
  if (threadIdx.x == 0) O[bc] = red[0] / (float)P;
}

// attention MLP: gap[8,256] -> 16 -> 256 -> sigmoid
__global__ void k_att(const float* __restrict__ gap, const float* __restrict__ w1,
                      const float* __restrict__ w2, float* __restrict__ av) {
  int b = blockIdx.x;
  __shared__ float a[256], h[16];
  a[threadIdx.x] = gap[b * 256 + threadIdx.x];
  __syncthreads();
  if (threadIdx.x < 16) {
    float s = 0.f;
    for (int c = 0; c < 256; ++c) s += w1[threadIdx.x * 256 + c] * a[c];
    h[threadIdx.x] = s;
  }
  __syncthreads();
  float s = 0.f;
  for (int k = 0; k < 16; ++k) s += w2[threadIdx.x * 16 + k] * h[k];
  av[b * 256 + threadIdx.x] = 1.f / (1.f + expf(-s));
}

__global__ void k_outer(const float* __restrict__ av, float* __restrict__ satt) {
  long long idx = (long long)blockIdx.x * 256 + threadIdx.x;
  int b = (int)(idx >> 16);
  int i = (int)((idx >> 8) & 255);
  int j = (int)(idx & 255);
  satt[idx] = av[b * 256 + i] * av[b * 256 + j];
}

// grid: (4, Bn*256); P = 784
__global__ void k_center(const float* __restrict__ X, const float* __restrict__ mu,
                         float* __restrict__ O, int P) {
  int i = blockIdx.x * 256 + threadIdx.x;
  if (i >= P) return;
  int bc = blockIdx.y;
  long long idx = (long long)bc * P + i;
  O[idx] = X[idx] - mu[bc];
}

__global__ void k_trace(const float* __restrict__ M, float* __restrict__ tr) {
  int b = blockIdx.x;
  __shared__ float red[256];
  red[threadIdx.x] = M[(long long)b * 65536 + threadIdx.x * 257];
  __syncthreads();
  for (int s = 128; s > 0; s >>= 1) {
    if (threadIdx.x < s) red[threadIdx.x] += red[threadIdx.x + s];
    __syncthreads();
  }
  if (threadIdx.x == 0) tr[b] = red[0];
}

// ilrmsn calls 1+2 closed form: Y2 = (A/tr)*(diag?0.75^5:1)*(1-0.25*(1-s_att))^5
__global__ void k_call12(const float* __restrict__ A, const float* __restrict__ satt,
                         const float* __restrict__ tr1, float* __restrict__ Y) {
  long long idx = (long long)blockIdx.x * 256 + threadIdx.x;
  int b = (int)(idx >> 16);
  int r = (int)((idx >> 8) & 255);
  int c = (int)(idx & 255);
  float f = 1.f - 0.25f * (1.f - satt[idx]);
  float f2 = f * f, f4 = f2 * f2, f5 = f4 * f;
  float v = A[idx] / tr1[b];
  if (r == c) v *= 0.2373046875f;  // 0.75^5
  Y[idx] = v * f5;
}

__global__ void k_axpby_diag(const float* __restrict__ X, float* __restrict__ Y,
                             float a, float b) {
  long long idx = (long long)blockIdx.x * 256 + threadIdx.x;
  int r = (int)((idx >> 8) & 255);
  int c = (int)(idx & 255);
  Y[idx] = a * X[idx] + (r == c ? b : 0.f);
}

__global__ void k_scale_sqrt(const float* __restrict__ Y, const float* __restrict__ tr,
                             float* __restrict__ O) {
  long long idx = (long long)blockIdx.x * 256 + threadIdx.x;
  int b = (int)(idx >> 16);
  O[idx] = Y[idx] * sqrtf(tr[b]);
}

// FC over lower-tri features
__global__ void k_fc(const float* __restrict__ A, const float* __restrict__ W,
                     const float* __restrict__ bias, float* __restrict__ cls) {
  int b = blockIdx.x, o = blockIdx.y;
  const float* Ab = A + (long long)b * 65536;
  const float* Wo = W + (long long)o * 32896;
  float acc = 0.f;
  for (int t = threadIdx.x; t < 32896; t += blockDim.x) {
    int i = (int)((sqrtf(8.f * t + 1.f) - 1.f) * 0.5f);
    while ((i + 1) * (i + 2) / 2 <= t) ++i;
    while (i * (i + 1) / 2 > t) --i;
    int j = t - i * (i + 1) / 2;
    acc += Wo[t] * Ab[i * 256 + j];
  }
  __shared__ float red[256];
  red[threadIdx.x] = acc; __syncthreads();
  for (int s = 128; s > 0; s >>= 1) {
    if (threadIdx.x < s) red[threadIdx.x] += red[threadIdx.x + s];
    __syncthreads();
  }
  if (threadIdx.x == 0) cls[b * 200 + o] = red[0] + bias[o];
}

// ---------------------------------------------------------------------------
// Host orchestration
// ---------------------------------------------------------------------------
struct BlkIdx { int c1, g1, b1, c2, g2, b2, dsc, dsg, dsb; };

extern "C" void kernel_launch(void* const* d_in, const int* in_sizes, int n_in,
                              void* d_out, int out_size, void* d_ws, size_t ws_size,
                              hipStream_t stream) {
  (void)in_sizes; (void)n_in; (void)out_size; (void)ws_size;
  auto in = [&](int i) { return (const float*)d_in[i]; };

  char* wp = (char*)d_ws;
  auto allocf = [&](size_t elems) -> float* {
    float* p = (float*)wp;
    size_t bytes = (elems * sizeof(float) + 255) & ~(size_t)255;
    wp += bytes;
    return p;
  };
  const int Bn = 8;
  float* big = allocf(25690112);                     // 8*64*224*224
  float* regs[4];
  for (int i = 0; i < 4; ++i) regs[i] = allocf(6422528);
  float* xc = allocf(1605632);                       // 8*256*784
  float* ns[6];
  for (int i = 0; i < 6; ++i) ns[i] = allocf(524288);
  unsigned short* abf = (unsigned short*)allocf(3211264);  // 6422528 bf16 (max activation)
  unsigned short* wbf = (unsigned short*)allocf(1179648);  // 2359296 bf16 (max weight)
  float* meanB = allocf(512);
  float* rstdB = allocf(512);
  float* gap = allocf(2048);
  float* av  = allocf(2048);
  float* tr1 = allocf(8);
  float* tr2 = allocf(8);

  float* cls_out  = (float*)d_out;
  float* A_out    = cls_out + 1600;
  float* satt_out = A_out + 524288;

  auto cvt = [&](const float* src, unsigned short* dst, long long total) {
    k_f2bf<<<(unsigned)((total + 255) / 256), 256, 0, stream>>>(src, dst, total);
  };
  auto conv = [&](const unsigned short* Xb, int widx, float* O,
                  int IC, int IH, int IW, int OC, int OH, int OW,
                  int KH, int KW, int st, int pad) {
    cvt(in(widx), wbf, (long long)OC * IC * KH * KW);
    dim3 g((unsigned)((Bn * OH * OW + 63) / 64), (unsigned)((OC + 63) / 64), 1);
    k_conv_wmma<<<g, 128, 0, stream>>>(Xb, wbf, O, Bn, IC, IH, IW, OC, OH, OW,
                                       KH, KW, st, pad,
                                       1.f / (float)(KH * KW), 1.f / (float)KW,
                                       1.f / (float)(OH * OW), 1.f / (float)OW);
  };
  auto bnrun = [&](float* X, const float* g_, const float* b_, const float* resid,
                   int relu, int C, int H, int W) {
    int P = H * W;
    k_bn_stats<<<C, 256, 0, stream>>>(X, meanB, rstdB, Bn, C, P);
    dim3 g((unsigned)((P + 255) / 256), (unsigned)(Bn * C), 1);
    k_bn_apply<<<g, 256, 0, stream>>>(X, X, meanB, rstdB, g_, b_, resid, C, P, relu);
  };
  auto mm = [&](const float* A, const float* B, float* C, int M, int N, int K,
                long long sA, long long sB, long long sC, float alpha, float beta, int tB) {
    dim3 g((unsigned)(N / 64), (unsigned)(M / 64), 8);
    k_mm<<<g, 128, 0, stream>>>(A, B, C, M, N, K, sA, sB, sC, alpha, beta, tB);
  };

  // ---- stem ----
  cvt(in(0), abf, 4816896);
  conv(abf, 1, big, 3, 448, 448, 64, 224, 224, 7, 7, 2, 3);
  bnrun(big, in(2), in(3), nullptr, 1, 64, 224, 224);
  {
    dim3 g((unsigned)((112 * 112 + 255) / 256), (unsigned)(Bn * 64), 1);
    k_maxpool<<<g, 256, 0, stream>>>(big, regs[0], 224, 224, 112, 112, 1.f / 112.f);
  }

  // ---- residual blocks ----
  const BlkIdx bidx[8] = {
    { 4, 5, 6, 7, 8, 9, -1, -1, -1},
    {10,11,12,13,14,15, -1, -1, -1},
    {16,17,18,19,20,21, 22, 23, 24},
    {25,26,27,28,29,30, -1, -1, -1},
    {31,32,33,34,35,36, 37, 38, 39},
    {40,41,42,43,44,45, -1, -1, -1},
    {46,47,48,49,50,51, 52, 53, 54},
    {55,56,57,58,59,60, -1, -1, -1},
  };
  const int binC[8]  = { 64, 64, 64,128,128,256,256,512};
  const int boutC[8] = { 64, 64,128,128,256,256,512,512};
  const int binH[8]  = {112,112,112, 56, 56, 28, 28, 28};
  const int bst[8]   = {  1,  1,  2,  1,  2,  1,  1,  1};

  int slot_cur = 0;
  for (int bi = 0; bi < 8; ++bi) {
    int IC = binC[bi], OC = boutC[bi], IH = binH[bi], st = bst[bi];
    int OH = (IH - 1) / st + 1;
    int s1 = -1, s2 = -1, s3 = -1;
    for (int s = 0; s < 4; ++s)
      if (s != slot_cur) { if (s1 < 0) s1 = s; else if (s2 < 0) s2 = s; else s3 = s; }
    float* cur = regs[slot_cur];
    float* t1 = regs[s1];
    float* t2 = regs[s2];
    float* t3 = regs[s3];
    const BlkIdx& X = bidx[bi];

    cvt(cur, abf, (long long)Bn * IC * IH * IH);       // shared by conv1 + ds conv
    conv(abf, X.c1, t1, IC, IH, IH, OC, OH, OH, 3, 3, st, 1);
    bnrun(t1, in(X.g1), in(X.b1), nullptr, 1, OC, OH, OH);

    const float* sc = cur;
    if (X.dsc >= 0) {
      conv(abf, X.dsc, t3, IC, IH, IH, OC, OH, OH, 1, 1, st, 0);
      bnrun(t3, in(X.dsg), in(X.dsb), nullptr, 0, OC, OH, OH);
      sc = t3;
    }
    cvt(t1, abf, (long long)Bn * OC * OH * OH);
    conv(abf, X.c2, t2, OC, OH, OH, OC, OH, OH, 3, 3, 1, 1);
    bnrun(t2, in(X.g2), in(X.b2), sc, 1, OC, OH, OH);  // fused BN + residual + relu
    slot_cur = s2;
  }

  // ---- projection to 256 ----
  float* cur = regs[slot_cur];
  int sp = (slot_cur == 0) ? 1 : 0;
  float* pro = regs[sp];
  cvt(cur, abf, (long long)Bn * 512 * 28 * 28);
  conv(abf, 61, pro, 512, 28, 28, 256, 28, 28, 1, 1, 1, 0);
  bnrun(pro, in(62), in(63), nullptr, 1, 256, 28, 28);

  // ---- attention + s_att ----
  k_gap<<<8 * 256, 128, 0, stream>>>(pro, gap, 784);
  k_att<<<8, 256, 0, stream>>>(gap, in(64), in(65), av);
  k_outer<<<2048, 256, 0, stream>>>(av, satt_out);

  // ---- covariance pooling (f32 WMMA) ----
  {
    dim3 g((unsigned)((784 + 255) / 256), (unsigned)(Bn * 256), 1);
    k_center<<<g, 256, 0, stream>>>(pro, gap, xc, 784);
  }
  mm(xc, xc, ns[0], 256, 256, 784, 200704, 200704, 65536, 1.f / 784.f, 0.f, 1);

  // ---- ilrmsn calls 1+2 (closed form) ----
  k_trace<<<8, 256, 0, stream>>>(ns[0], tr1);
  k_call12<<<2048, 256, 0, stream>>>(ns[0], satt_out, tr1, ns[1]);
  k_trace<<<8, 256, 0, stream>>>(ns[1], tr2);

  // ---- ilrmsn call 3: Newton-Schulz, 5 iterations (f32 WMMA) ----
  k_axpby_diag<<<2048, 256, 0, stream>>>(ns[1], ns[2], -0.5f, 1.5f);
  mm(ns[1], ns[2], ns[3], 256, 256, 256, 65536, 65536, 65536, 1.f, 0.f, 0);
  float* yp = ns[3];
  float* zp = ns[2];
  float* freeb[4] = {ns[0], ns[1], ns[4], ns[5]};
  for (int it = 1; it <= 3; ++it) {
    float* zy = freeb[0]; float* yn = freeb[1]; float* zn = freeb[2];
    mm(zp, yp, zy, 256, 256, 256, 65536, 65536, 65536, -0.5f, 1.5f, 0);
    mm(yp, zy, yn, 256, 256, 256, 65536, 65536, 65536, 1.f, 0.f, 0);
    mm(zy, zp, zn, 256, 256, 256, 65536, 65536, 65536, 1.f, 0.f, 0);
    float* oy = yp; float* oz = zp;
    yp = yn; zp = zn;
    freeb[0] = oy; freeb[1] = oz; freeb[2] = zy;
  }
  {
    float* zy = freeb[0]; float* yf = freeb[1];
    mm(zp, yp, zy, 256, 256, 256, 65536, 65536, 65536, -0.5f, 1.5f, 0);
    mm(yp, zy, yf, 256, 256, 256, 65536, 65536, 65536, 1.f, 0.f, 0);
    k_scale_sqrt<<<2048, 256, 0, stream>>>(yf, tr2, A_out);
  }

  // ---- classifier ----
  k_fc<<<dim3(8, 200), 256, 0, stream>>>(A_out, in(66), in(67), cls_out);
}
